// NoveltyDetector_10746008175308
// MI455X (gfx1250) — matrix operand order repaired and checked by reading, not compiled, not run on hardware
//
#include <hip/hip_runtime.h>
#include <hip/hip_bf16.h>
#include <math.h>

// ---------------------------------------------------------------------------
// Novelty detector for MI455X (gfx1250, wave32, WMMA).
//
// Distance GEMM: 52.4 GFLOP vs ~52 MB unique bytes (fits in 192MB L2) ->
// compute bound -> bf16 WMMA 16x16x32 (16 KFLOP/instr), fp32 accumulate.
// Norms are fp32 of the BF16-ROUNDED vectors so sq >= 0 stays consistent.
// Top-k ranks on the monotone proxy (||m||^2 - 2 m.e); sqrt/+||e||^2 at merge.
// Epilogue is branchless (cmp/max/cndmask) so it co-executes under the XDL
// pipe instead of serializing on exec-mask branch chains.
// ---------------------------------------------------------------------------

typedef __bf16 bf16_t;
typedef __attribute__((ext_vector_type(16))) __bf16 v16bf;
typedef __attribute__((ext_vector_type(8)))  __bf16 v8bf;
typedef __attribute__((ext_vector_type(8)))  float  v8f;
typedef __attribute__((ext_vector_type(4)))  float  v4f;

#define B_ROWS      2048
#define DIM         128
#define C_ROWS      100000
#define NCHUNK      8
#define CHUNK_ROWS  12800       // 50 iters * 256 rows; 8*12800 = 102400 padded
#define C_PAD       (NCHUNK * CHUNK_ROWS)
#define ITERS       50
#define KNEAR       10
#define BIGF        3.0e38f

// ---------------------------------------------------------------------------
// Encoder stage 1: h = relu(state @ W1 + b1).  Tiny (67 MFLOP) -> plain VALU.
// ---------------------------------------------------------------------------
__global__ __launch_bounds__(128) void nd_encoder1(
    const float* __restrict__ state, const float* __restrict__ W1,
    const float* __restrict__ b1, float* __restrict__ h) {
  __shared__ float s[16 * DIM];
  const int t = threadIdx.x;          // 0..127 = output column
  const int row0 = blockIdx.x * 16;
  for (int r = 0; r < 16; ++r) s[r * DIM + t] = state[(row0 + r) * DIM + t];
  __syncthreads();
  float acc[16];
  const float bias = b1[t];
  #pragma unroll
  for (int r = 0; r < 16; ++r) acc[r] = bias;
  for (int d = 0; d < DIM; ++d) {
    const float w = W1[d * DIM + t];
    #pragma unroll
    for (int r = 0; r < 16; ++r) acc[r] = fmaf(s[r * DIM + d], w, acc[r]);
  }
  for (int r = 0; r < 16; ++r)
    h[(row0 + r) * DIM + t] = fmaxf(acc[r], 0.0f);
}

// ---------------------------------------------------------------------------
// Encoder stage 2: enc = h @ W2 + b2 -> encT bf16 [K=128][row=2048] + fp32
// ||enc_bf16||^2 per row (deterministic serial reduce).
// ---------------------------------------------------------------------------
__global__ __launch_bounds__(128) void nd_encoder2(
    const float* __restrict__ h, const float* __restrict__ W2,
    const float* __restrict__ b2, bf16_t* __restrict__ encT,
    float* __restrict__ sqe) {
  __shared__ float s[16 * DIM];
  const int t = threadIdx.x;
  const int row0 = blockIdx.x * 16;
  for (int r = 0; r < 16; ++r) s[r * DIM + t] = h[(row0 + r) * DIM + t];
  __syncthreads();
  float acc[16];
  const float bias = b2[t];
  #pragma unroll
  for (int r = 0; r < 16; ++r) acc[r] = bias;
  for (int d = 0; d < DIM; ++d) {
    const float w = W2[d * DIM + t];
    #pragma unroll
    for (int r = 0; r < 16; ++r) acc[r] = fmaf(s[r * DIM + d], w, acc[r]);
  }
  __syncthreads();                     // done reading s; reuse it for squares
  for (int r = 0; r < 16; ++r) {
    const bf16_t bv = (bf16_t)acc[r];
    encT[(size_t)t * B_ROWS + row0 + r] = bv;   // transposed store
    const float vf = (float)bv;                  // bf16-rounded value
    s[r * DIM + t] = vf * vf;
  }
  __syncthreads();
  if (t < 16) {
    float sum = 0.0f;
    for (int d = 0; d < DIM; ++d) sum += s[t * DIM + d];
    sqe[row0 + t] = sum;
  }
}

// ---------------------------------------------------------------------------
// Memory prep: bf16 convert (padded to C_PAD rows) + fp32 norms of the
// bf16-rounded rows.  Pad rows -> data 0, norm +BIG so they never rank.
// ---------------------------------------------------------------------------
__global__ __launch_bounds__(256) void nd_prep_mem(
    const float* __restrict__ mem, bf16_t* __restrict__ memb,
    float* __restrict__ sqm) {
  __shared__ float red[256];
  const int t = threadIdx.x;
  const int row = blockIdx.x * 2 + (t >> 7);
  const int c = t & 127;
  float v = 0.0f;
  if (row < C_ROWS) v = mem[(size_t)row * DIM + c];
  const bf16_t bv = (bf16_t)v;
  memb[(size_t)row * DIM + c] = (row < C_ROWS) ? bv : (bf16_t)0.0f;
  const float vf = (float)bv;
  red[t] = (row < C_ROWS) ? vf * vf : 0.0f;
  __syncthreads();
  for (int off = 64; off > 0; off >>= 1) {
    if (c < off) red[t] += red[t + off];
    __syncthreads();
  }
  if (c == 0) sqm[row] = (row < C_ROWS) ? red[t] : BIGF;
}

// ---------------------------------------------------------------------------
// Branchless guarded insert into sorted-descending tk[] (tk[0] = worst kept).
// Accept path: 9x (cmp + max + cndmask) + 1 min, no exec-mask branch chain.
// Invariant: tk stays sorted descending; result = 10 smallest seen.
// ---------------------------------------------------------------------------
__device__ __forceinline__ void topk_insert(float (&tk)[KNEAR], float c) {
  if (c < tk[0]) {
    #pragma unroll
    for (int j = 0; j < KNEAR - 1; ++j)
      tk[j] = (c < tk[j]) ? fmaxf(c, tk[j + 1]) : tk[j];
    tk[KNEAR - 1] = fminf(tk[KNEAR - 1], c);
  }
}

// ---------------------------------------------------------------------------
// Main fused kernel: D = memory_bf16 @ encT.  Per iteration each wave runs
// TWO independent 16x16 tiles (two WMMA accumulator chains sharing the
// B fragments) -> 8 interleaved WMMAs hide XDL latency, and each lane sees
// 800 candidates per kept-10 (lower accept rate -> cheaper epilogue).
// A min-tree over the 16 candidates gives one wave-level skip guard.
// C/D layout: lane -> enc row = n0 + (lane&15); VGPR r -> M = m0+r+8*(lane>>4).
// ---------------------------------------------------------------------------
__global__ __launch_bounds__(256) void nd_dist_topk(
    const bf16_t* __restrict__ memb, const float* __restrict__ sqm,
    const bf16_t* __restrict__ encT, float* __restrict__ partial) {
  __shared__ float ls[256 * KNEAR];
  const int chunk = blockIdx.x;                 // 0..7
  const int ntile = blockIdx.y;                 // 0..127
  const int tid  = threadIdx.x;
  const int lane = tid & 31;
  const int wave = tid >> 5;
  const int hf   = lane >> 4;                   // lane half
  const int lr   = lane & 15;
  const int n0   = ntile * 16;

  // B fragments (32x16 bf16): lane holds N = n0 + lr, K runs [32kk+8hf,+8) and [+16,+8).
  // Loaded once, reused by both tiles for all 50 iterations.
  v16bf bfr[4];
  #pragma unroll
  for (int kk = 0; kk < 4; ++kk) {
    #pragma unroll
    for (int s = 0; s < 16; ++s) {
      const int k = 32 * kk + ((s < 8) ? (8 * hf + s) : (16 + 8 * hf + (s - 8)));
      bfr[kk][s] = encT[(size_t)k * B_ROWS + n0 + lr];
    }
  }
  const v16bf bf0 = bfr[0];
  const v16bf bf1 = bfr[1];
  const v16bf bf2 = bfr[2];
  const v16bf bf3 = bfr[3];

  float tk[KNEAR];
  #pragma unroll
  for (int j = 0; j < KNEAR; ++j) tk[j] = BIGF;

  const int base = chunk * CHUNK_ROWS + wave * 32;   // wave's 32-row slab
  for (int it = 0; it < ITERS; ++it) {
    const bf16_t* rp0 = memb + (size_t)(base + it * 256 + lr) * DIM;
    const bf16_t* rp1 = rp0 + 16 * DIM;
    if (it + 1 < ITERS) {
      __builtin_prefetch(rp0 + 256 * DIM, 0, 3);
      __builtin_prefetch(rp1 + 256 * DIM, 0, 3);
    }

    // A fragments (16x32 bf16): lane holds K runs [32kk+8hf,+8) and [+16,+8)
    v16bf a0[4], a1[4];
    #pragma unroll
    for (int kk = 0; kk < 4; ++kk) {
      const v8bf lo0 = *(const v8bf*)(rp0 + 32 * kk + 8 * hf);
      const v8bf hi0 = *(const v8bf*)(rp0 + 32 * kk + 16 + 8 * hf);
      const v8bf lo1 = *(const v8bf*)(rp1 + 32 * kk + 8 * hf);
      const v8bf hi1 = *(const v8bf*)(rp1 + 32 * kk + 16 + 8 * hf);
      v16bf av0, av1;
      #pragma unroll
      for (int e = 0; e < 8; ++e) {
        av0[e] = lo0[e]; av0[e + 8] = hi0[e];
        av1[e] = lo1[e]; av1[e + 8] = hi1[e];
      }
      a0[kk] = av0; a1[kk] = av1;
    }

    v8f c0 = {}, c1 = {};
    c0 = __builtin_amdgcn_wmma_f32_16x16x32_bf16(false, a0[0], false, bf0, (short)0, c0, false, false);
    c1 = __builtin_amdgcn_wmma_f32_16x16x32_bf16(false, a1[0], false, bf0, (short)0, c1, false, false);
    c0 = __builtin_amdgcn_wmma_f32_16x16x32_bf16(false, a0[1], false, bf1, (short)0, c0, false, false);
    c1 = __builtin_amdgcn_wmma_f32_16x16x32_bf16(false, a1[1], false, bf1, (short)0, c1, false, false);
    c0 = __builtin_amdgcn_wmma_f32_16x16x32_bf16(false, a0[2], false, bf2, (short)0, c0, false, false);
    c1 = __builtin_amdgcn_wmma_f32_16x16x32_bf16(false, a1[2], false, bf2, (short)0, c1, false, false);
    c0 = __builtin_amdgcn_wmma_f32_16x16x32_bf16(false, a0[3], false, bf3, (short)0, c0, false, false);
    c1 = __builtin_amdgcn_wmma_f32_16x16x32_bf16(false, a1[3], false, bf3, (short)0, c1, false, false);

    // epilogue: cand = ||m||^2 - 2*dot  (monotone proxy for distance)
    const int mb = chunk * CHUNK_ROWS + it * 256 + wave * 32 + 8 * hf;
    const v4f q0 = *(const v4f*)(sqm + mb);
    const v4f q1 = *(const v4f*)(sqm + mb + 4);
    const v4f q2 = *(const v4f*)(sqm + mb + 16);
    const v4f q3 = *(const v4f*)(sqm + mb + 20);

    float cand[16];
    #pragma unroll
    for (int r = 0; r < 8; ++r) {
      cand[r]     = fmaf(-2.0f, c0[r], (r < 4) ? q0[r] : q1[r - 4]);
      cand[r + 8] = fmaf(-2.0f, c1[r], (r < 4) ? q2[r] : q3[r - 4]);
    }
    float m = cand[0];
    #pragma unroll
    for (int i = 1; i < 16; ++i) m = fminf(m, cand[i]);
    if (m < tk[0]) {                   // wave-level skip when nothing can land
      #pragma unroll
      for (int i = 0; i < 16; ++i) topk_insert(tk, cand[i]);
    }
  }

  // merge 16 lane-lists per enc row (2 halves x 8 waves) through LDS
  #pragma unroll
  for (int j = 0; j < KNEAR; ++j) ls[tid * KNEAR + j] = tk[j];
  __syncthreads();
  if (tid < 16) {
    float best[KNEAR];
    #pragma unroll
    for (int j = 0; j < KNEAR; ++j) best[j] = BIGF;
    for (int w = 0; w < 8; ++w)
      for (int half = 0; half < 2; ++half) {
        const int src = (w * 32 + half * 16 + tid) * KNEAR;
        // lists sorted descending -> scan from the smallest, early out
        for (int j = KNEAR - 1; j >= 0; --j) {
          const float cnd = ls[src + j];
          if (cnd >= best[0]) break;
          topk_insert(best, cnd);
        }
      }
    float* dst = partial + ((size_t)(n0 + tid) * NCHUNK + chunk) * KNEAR;
    #pragma unroll
    for (int j = 0; j < KNEAR; ++j) dst[j] = best[j];
  }
}

// ---------------------------------------------------------------------------
// Final merge: per enc row, select 10 smallest of 8 chunk-lists (80 vals),
// add ||enc||^2, sqrt, mean.
// ---------------------------------------------------------------------------
__global__ __launch_bounds__(256) void nd_merge(
    const float* __restrict__ partial, const float* __restrict__ sqe,
    float* __restrict__ out) {
  const int row = blockIdx.x * blockDim.x + threadIdx.x;
  if (row >= B_ROWS) return;
  const float* p = partial + (size_t)row * NCHUNK * KNEAR;
  float best[KNEAR];
  #pragma unroll
  for (int j = 0; j < KNEAR; ++j) best[j] = BIGF;
  for (int i = 0; i < NCHUNK * KNEAR; ++i) topk_insert(best, p[i]);
  const float se = sqe[row];
  float sum = 0.0f;
  #pragma unroll
  for (int j = 0; j < KNEAR; ++j)
    sum += sqrtf(fmaxf(best[j] + se, 1e-12f));
  out[row] = sum * (1.0f / KNEAR);
}

// ---------------------------------------------------------------------------
// Launch.  Workspace layout (bytes, 256-aligned):
//   h       : 2048*128*4      = 1,048,576   @ 0
//   encT    : 128*2048*2      =   524,288   @ 1,048,576
//   sqe     : 2048*4          =     8,192   @ 1,572,864
//   sqm     : 102400*4        =   409,600   @ 1,581,056
//   partial : 2048*8*10*4     =   655,360   @ 1,990,656
//   memb    : 102400*128*2    = 26,214,400  @ 2,646,016   (total ~28.9 MB)
// ---------------------------------------------------------------------------
extern "C" void kernel_launch(void* const* d_in, const int* in_sizes, int n_in,
                              void* d_out, int out_size, void* d_ws, size_t ws_size,
                              hipStream_t stream) {
  const float* state = (const float*)d_in[0];
  const float* W1    = (const float*)d_in[1];
  const float* b1    = (const float*)d_in[2];
  const float* W2    = (const float*)d_in[3];
  const float* b2    = (const float*)d_in[4];
  const float* mem   = (const float*)d_in[5];
  float* out = (float*)d_out;

  char* ws = (char*)d_ws;
  float*  h       = (float*)(ws);
  bf16_t* encT    = (bf16_t*)(ws + 1048576);
  float*  sqe     = (float*)(ws + 1572864);
  float*  sqm     = (float*)(ws + 1581056);
  float*  partial = (float*)(ws + 1990656);
  bf16_t* memb    = (bf16_t*)(ws + 2646016);

  nd_encoder1<<<dim3(B_ROWS / 16), dim3(128), 0, stream>>>(state, W1, b1, h);
  nd_encoder2<<<dim3(B_ROWS / 16), dim3(128), 0, stream>>>(h, W2, b2, encT, sqe);
  nd_prep_mem<<<dim3(C_PAD / 2), dim3(256), 0, stream>>>(mem, memb, sqm);
  nd_dist_topk<<<dim3(NCHUNK, B_ROWS / 16), dim3(256), 0, stream>>>(memb, sqm, encT, partial);
  nd_merge<<<dim3(B_ROWS / 256), dim3(256), 0, stream>>>(partial, sqe, out);
}
